// LSTM_Model_3_1949915153231
// MI455X (gfx1250) — 16-bit and 8-bit matrix operand order repaired and checked by reading, compiled, not run on hardware
//
#include <hip/hip_runtime.h>
#include <math.h>


typedef _Float16 h8   __attribute__((ext_vector_type(8)));
typedef _Float16 v16h __attribute__((ext_vector_type(16)));
typedef float    v8f  __attribute__((ext_vector_type(8)));
typedef int      v2i  __attribute__((ext_vector_type(2)));
typedef int      v4i  __attribute__((ext_vector_type(4)));
typedef int      v8i  __attribute__((ext_vector_type(8)));

#define B_TOT 64
#define T_LEN 2048
#define F_IN  64
#define UNITS 128
#define GDIM  512
#define BW    16
#define SX1   72
#define SH    136
#define SX2   136

#define LOG2E  1.4426950408889634f

__device__ __forceinline__ float fexp2(float x) {
#if __has_builtin(__builtin_amdgcn_exp2f)
    return __builtin_amdgcn_exp2f(x);
#else
    return exp2f(x);
#endif
}
__device__ __forceinline__ float frcp(float x) {
#if __has_builtin(__builtin_amdgcn_rcpf)
    return __builtin_amdgcn_rcpf(x);
#else
    return 1.0f / x;
#endif
}
__device__ __forceinline__ float sigf(float v) {
    return frcp(1.0f + fexp2(v * -LOG2E));
}
__device__ __forceinline__ float tanh_fast(float v) {
#if __has_builtin(__builtin_amdgcn_tanhf)
    return __builtin_amdgcn_tanhf(v);
#else
    return 1.0f - 2.0f * frcp(fexp2(v * (2.0f * LOG2E)) + 1.0f);
#endif
}

__device__ __forceinline__ unsigned char sw_e4m3(float f) {
    unsigned u = __float_as_uint(f);
    unsigned s = (u >> 31) << 7;
    float a = fabsf(f);
    if (!(a < 448.0f)) return (unsigned char)(s | 0x7E);
    if (a < 0.0009765625f) return (unsigned char)s;
    int ex; float m = frexpf(a, &ex);
    int E = ex - 1;
    unsigned code;
    if (E < -6) {
        int d = (int)rintf(a * 512.0f);
        code = (d >= 8) ? 0x08u : (unsigned)d;
    } else {
        int q = (int)rintf((m * 2.0f - 1.0f) * 8.0f);
        if (q == 8) { E += 1; q = 0; }
        code = (E > 8) ? 0x7Eu : (unsigned)(((E + 7) << 3) | q);
    }
    return (unsigned char)(s | code);
}

__device__ __forceinline__ unsigned pk4_fp8(float a, float b, float c, float d) {
#if __has_builtin(__builtin_amdgcn_cvt_pk_fp8_f32)
    int v = __builtin_amdgcn_cvt_pk_fp8_f32(a, b, 0, false);
    v = __builtin_amdgcn_cvt_pk_fp8_f32(c, d, v, true);
    return (unsigned)v;
#else
    return (unsigned)sw_e4m3(a) | ((unsigned)sw_e4m3(b) << 8) |
           ((unsigned)sw_e4m3(c) << 16) | ((unsigned)sw_e4m3(d) << 24);
#endif
}

__device__ __forceinline__ unsigned char cv_fp8(float a) {
#if __has_builtin(__builtin_amdgcn_cvt_pk_fp8_f32)
    return (unsigned char)(__builtin_amdgcn_cvt_pk_fp8_f32(a, 0.0f, 0, false) & 0xFF);
#else
    return sw_e4m3(a);
#endif
}

__device__ __forceinline__ v16h loadA16(const _Float16* p) {
    const h8 a0 = *(const h8*)p;
    const h8 a1 = *(const h8*)(p + 16);
    return __builtin_shufflevector(a0, a1,
        0, 1, 2, 3, 4, 5, 6, 7, 8, 9, 10, 11, 12, 13, 14, 15);
}
__device__ __forceinline__ v8i loadA8(const unsigned char* p) {
    const v2i c0 = *(const v2i*)p;
    const v2i c1 = *(const v2i*)(p + 16);
    const v2i c2 = *(const v2i*)(p + 32);
    const v2i c3 = *(const v2i*)(p + 48);
    const v4i lo = __builtin_shufflevector(c0, c1, 0, 1, 2, 3);
    const v4i hi = __builtin_shufflevector(c2, c3, 0, 1, 2, 3);
    return __builtin_shufflevector(lo, hi, 0, 1, 2, 3, 4, 5, 6, 7);
}

#define WMMA16(A, B, C) \
    __builtin_amdgcn_wmma_f32_16x16x32_f16(false, A, false, B, (short)0, C, false, false)
#define WMMA8(A, B, C) \
    __builtin_amdgcn_wmma_f32_16x16x64_fp8_fp8(A, B, (short)0, C, false, false)

__global__ __launch_bounds__(256, 1)
void lstm2_persistent(const float* __restrict__ x,
                      const float* __restrict__ W1, const float* __restrict__ U1,
                      const float* __restrict__ b1,
                      const float* __restrict__ W2, const float* __restrict__ U2,
                      const float* __restrict__ b2,
                      float* __restrict__ out) {
    __shared__ unsigned char As1x[BW * SX1];
    __shared__ _Float16     As1h[BW * SH];
    __shared__ unsigned char As2x[BW * SX2];
    __shared__ _Float16     As2h[BW * SH];

    const int tid   = threadIdx.x;
    const int w     = tid >> 5;
    const int lane  = tid & 31;
    const int lmod  = lane & 15;
    const int lhalf = lane >> 4;
    const int b0    = blockIdx.x * BW;
    const int col   = w * 16 + lmod;

    v8i fW1[4];
    #pragma unroll
    for (int g = 0; g < 4; ++g) {
        const int n = g * UNITS + col;
        v8i f;
        #pragma unroll
        for (int d = 0; d < 8; ++d) {
            const int ks = 16 * (d >> 1) + lhalf * 8 + 4 * (d & 1);
            f[d] = (int)pk4_fp8(W1[(ks + 0) * GDIM + n], W1[(ks + 1) * GDIM + n],
                                W1[(ks + 2) * GDIM + n], W1[(ks + 3) * GDIM + n]);
        }
        fW1[g] = f;
    }
    v8i fW2[2][4];
    #pragma unroll
    for (int kc = 0; kc < 2; ++kc) {
        #pragma unroll
        for (int g = 0; g < 4; ++g) {
            const int n = g * UNITS + col;
            v8i f;
            #pragma unroll
            for (int d = 0; d < 8; ++d) {
                const int ks = kc * 64 + (16 * (d >> 1) + lhalf * 8 + 4 * (d & 1));
                f[d] = (int)pk4_fp8(W2[(ks + 0) * GDIM + n], W2[(ks + 1) * GDIM + n],
                                    W2[(ks + 2) * GDIM + n], W2[(ks + 3) * GDIM + n]);
            }
            fW2[kc][g] = f;
        }
    }
    v16h fU1[4][4], fU2[4][4];
    #pragma unroll
    for (int kc = 0; kc < 4; ++kc) {
        #pragma unroll
        for (int g = 0; g < 4; ++g) {
            const int n = g * UNITS + col;
            v16h f1, f2;
            #pragma unroll
            for (int q = 0; q < 16; ++q) {
                const int k = kc * 32 + lhalf * 8 + (q & 7) + ((q >> 3) << 4);
                f1[q] = (_Float16)U1[k * GDIM + n];
                f2[q] = (_Float16)U2[k * GDIM + n];
            }
            fU1[kc][g] = f1;
            fU2[kc][g] = f2;
        }
    }

    float bi1[4], bi2[4];
    #pragma unroll
    for (int g = 0; g < 4; ++g) {
        bi1[g] = b1[g * UNITS + col];
        bi2[g] = b2[g * UNITS + col];
    }

    for (int i = tid; i < BW * SX1; i += 256) As1x[i] = 0;
    for (int i = tid; i < BW * SX2; i += 256) As2x[i] = 0;
    for (int i = tid; i < BW * SH; i += 256) { As1h[i] = (_Float16)0.0f; As2h[i] = (_Float16)0.0f; }

    v8f c1 = {};
    v8f c2 = {};

    __syncthreads();

    const _Float16*      a16p1 = &As1h[lmod * SH + lhalf * 8];
    const _Float16*      a16p2 = &As2h[lmod * SH + lhalf * 8];
    const unsigned char* a8p1  = &As1x[lmod * SX1 + lhalf * 8];
    const unsigned char* a8p2  = &As2x[lmod * SX2 + lhalf * 8];

    for (int t = 0; t < T_LEN; ++t) {
        {
            const int bb = tid >> 4;
            const int ff = (tid & 15) * 4;
            const float* xp = x + (size_t)(b0 + bb) * (T_LEN * F_IN)
                                + (size_t)t * F_IN + ff;
            const float4 xv = *(const float4*)xp;
            *(unsigned*)&As1x[bb * SX1 + ff] = pk4_fp8(xv.x, xv.y, xv.z, xv.w);
            if (t + 1 < T_LEN) __builtin_prefetch(xp + F_IN, 0, 1);
        }
        __syncthreads();

        v8f acc0 = {}, acc1 = {}, acc2 = {}, acc3 = {};
        {
            const v8i a = loadA8(a8p1);
            acc0 = WMMA8(a, fW1[0], acc0);
            acc1 = WMMA8(a, fW1[1], acc1);
            acc2 = WMMA8(a, fW1[2], acc2);
            acc3 = WMMA8(a, fW1[3], acc3);
        }
        #pragma unroll
        for (int kc = 0; kc < 4; ++kc) {
            const v16h a = loadA16(a16p1 + kc * 32);
            acc0 = WMMA16(a, fU1[kc][0], acc0);
            acc1 = WMMA16(a, fU1[kc][1], acc1);
            acc2 = WMMA16(a, fU1[kc][2], acc2);
            acc3 = WMMA16(a, fU1[kc][3], acc3);
        }
        __syncthreads();

        v8f hv;
        #pragma unroll
        for (int r = 0; r < 8; ++r) {
            const float iv = sigf(acc0[r] + bi1[0]);
            const float fv = sigf(acc1[r] + bi1[1]);
            const float gv = tanh_fast(acc2[r] + bi1[2]);
            const float ov = sigf(acc3[r] + bi1[3]);
            const float cn = fv * c1[r] + iv * gv;
            c1[r] = cn;
            hv[r] = ov * tanh_fast(cn);
        }
        #pragma unroll
        for (int r = 0; r < 8; ++r) {
            const int bb = lhalf * 8 + r;
            As1h[bb * SH + col]  = (_Float16)hv[r];
            As2x[bb * SX2 + col] = cv_fp8(hv[r]);
        }
        __syncthreads();

        acc0 = (v8f){}; acc1 = (v8f){}; acc2 = (v8f){}; acc3 = (v8f){};
        #pragma unroll
        for (int kc = 0; kc < 2; ++kc) {
            const v8i a = loadA8(a8p2 + kc * 64);
            acc0 = WMMA8(a, fW2[kc][0], acc0);
            acc1 = WMMA8(a, fW2[kc][1], acc1);
            acc2 = WMMA8(a, fW2[kc][2], acc2);
            acc3 = WMMA8(a, fW2[kc][3], acc3);
        }
        #pragma unroll
        for (int kc = 0; kc < 4; ++kc) {
            const v16h a = loadA16(a16p2 + kc * 32);
            acc0 = WMMA16(a, fU2[kc][0], acc0);
            acc1 = WMMA16(a, fU2[kc][1], acc1);
            acc2 = WMMA16(a, fU2[kc][2], acc2);
            acc3 = WMMA16(a, fU2[kc][3], acc3);
        }
        __syncthreads();

        #pragma unroll
        for (int r = 0; r < 8; ++r) {
            const float iv = sigf(acc0[r] + bi2[0]);
            const float fv = sigf(acc1[r] + bi2[1]);
            const float gv = tanh_fast(acc2[r] + bi2[2]);
            const float ov = sigf(acc3[r] + bi2[3]);
            const float cn = fv * c2[r] + iv * gv;
            c2[r] = cn;
            hv[r] = ov * tanh_fast(cn);
        }
        #pragma unroll
        for (int r = 0; r < 8; ++r) {
            const int bb = lhalf * 8 + r;
            As2h[bb * SH + col] = (_Float16)hv[r];
            if (t == T_LEN - 1) {
                out[(size_t)(b0 + bb) * UNITS + col] = hv[r];
            }
        }
    }
}

extern "C" void kernel_launch(void* const* d_in, const int* in_sizes, int n_in,
                              void* d_out, int out_size, void* d_ws, size_t ws_size,
                              hipStream_t stream) {
    (void)in_sizes; (void)n_in; (void)d_ws; (void)ws_size; (void)out_size;
    const float* x  = (const float*)d_in[0];
    const float* W1 = (const float*)d_in[1];
    const float* U1 = (const float*)d_in[2];
    const float* b1 = (const float*)d_in[3];
    const float* W2 = (const float*)d_in[4];
    const float* U2 = (const float*)d_in[5];
    const float* b2 = (const float*)d_in[6];
    float* out = (float*)d_out;

    lstm2_persistent<<<dim3(B_TOT / BW), dim3(256), 0, stream>>>(
        x, W1, U1, b1, W2, U2, b2, out);
}
